// EnHSG_52836687675886
// MI455X (gfx1250) — matrix operand order repaired and checked by reading, not compiled, not run on hardware
//
#include <hip/hip_runtime.h>

#define BN   50      // nodes
#define DIM  256     // hidden dim
#define TDIM 64      // time dim
#define NPAD 64      // padded N for 16x16 tiles
#define NEGV -9e15f

typedef _Float16 v16h __attribute__((ext_vector_type(16)));
typedef _Float16 h2   __attribute__((ext_vector_type(2)));
typedef float    v8f  __attribute__((ext_vector_type(8)));

#define HP  264   // s_h  row stride (halves), 64 rows x 256 used cols
#define HTP 72    // s_hT row stride (halves), 256 rows x 64 used cols
#define AP  68    // s_alpha row stride (floats)
#define AH  72    // s_ah row stride (halves)

#define SM_ALPHA (64*AP)
#define SM_TREL  (5*TDIM)
#define SM_W     (TDIM)
#define SM_H     (64*HP)
#define SM_HT    (256*HTP)
#define SM_AR    (5*DIM)
#define SM_AH    (64*AH)
#define SMEM_BYTES (4*(SM_ALPHA+SM_TREL+SM_W) + 2*(SM_H+SM_HT+SM_AR+SM_AH))

// f16 WMMA A-fragment K index: p<4 -> 8g+2p ; p>=4 -> 16+8g+2(p-4)
__device__ __forceinline__ int kA(int p, int g) { return ((p & 4) << 2) + (g << 3) + ((p & 3) << 1); }
// f16 WMMA B-fragment K index: same as A: p<4 -> 8g+2p ; p>=4 -> 16+8g+2(p-4)
__device__ __forceinline__ int kB(int p, int g) { return ((p & 4) << 2) + (g << 3) + ((p & 3) << 1); }

__global__ __launch_bounds__(256) void
tgat_fused_kernel(const float* __restrict__ hidden, const int* __restrict__ adj,
                  const float* __restrict__ A_interval, const float* __restrict__ a_rel,
                  const float* __restrict__ t_rel, const float* __restrict__ time_w,
                  float* __restrict__ out)
{
    extern __shared__ __align__(16) char smem[];
    float*    s_alpha = (float*)smem;              // [64][AP] logits -> exp
    float*    s_trel  = s_alpha + SM_ALPHA;        // [5][64]
    float*    s_w     = s_trel + SM_TREL;          // [64]
    _Float16* s_h     = (_Float16*)(s_w + SM_W);   // [64][HP]  h (f16), rows>=50 zero
    _Float16* s_hT    = s_h + SM_H;                // [256][HTP] h transposed (f16)
    _Float16* s_ar    = s_hT + SM_HT;              // [5][256]  a_rel (f16)
    _Float16* s_ah    = s_ar + SM_AR;              // [64][AH]  alpha (f16)

    const int b    = blockIdx.x;
    const int tid  = threadIdx.x;
    const int lane = tid & 31;
    const int wave = tid >> 5;
    const int g    = lane >> 4;    // lane group (0/1)
    const int ln   = lane & 15;

    // ---------- Phase 0: stage batch into LDS ----------
    const float* hb = hidden + (size_t)b * BN * DIM;
    for (int idx = tid; idx < NPAD * DIM; idx += 256) {
        int row = idx >> 8, col = idx & 255;
        float v = (row < BN) ? hb[row * DIM + col] : 0.f;
        _Float16 hv = (_Float16)v;
        s_h[row * HP + col]   = hv;
        s_hT[col * HTP + row] = hv;
    }
    for (int idx = tid; idx < 5 * DIM; idx += 256)  s_ar[idx]  = (_Float16)a_rel[idx];
    for (int idx = tid; idx < 5 * TDIM; idx += 256) s_trel[idx] = t_rel[idx];
    if (tid < TDIM) s_w[tid] = time_w[tid];
    __syncthreads();

    // ---------- Phase 1: structural WMMA (5 relations, fused select) + time encode ----------
    const int*   adjb = adj        + (size_t)b * BN * BN;
    const float* Aib  = A_interval + (size_t)b * BN * BN;

    for (int t = wave; t < 16; t += 8) {          // 4x4 tiles of the 64x64 pair grid
        int i0 = (t >> 2) << 4, j0 = (t & 3) << 4;
        int jj = j0 + ln;
        int adjv[8]; float Ai[8];
#pragma unroll
        for (int v = 0; v < 8; ++v) {
            int ii = i0 + v + (g << 3);
            bool valid = (ii < BN) && (jj < BN);
            adjv[v] = valid ? adjb[ii * BN + jj] : 0;
            Ai[v]   = valid ? Aib[ii * BN + jj]  : 0.f;
        }
        // B fragments (h rows j0+ln) hoisted across the 5 relations
        v16h bf[8];
#pragma unroll
        for (int s = 0; s < 8; ++s) {
#pragma unroll
            for (int p = 0; p < 8; ++p) {
                int kk = (s << 5) + kB(p, g);
                h2 hv = *(const h2*)&s_h[(j0 + ln) * HP + kk];
                bf[s][2*p] = hv.x; bf[s][2*p+1] = hv.y;
            }
        }
        float selacc[8];
#pragma unroll
        for (int v = 0; v < 8; ++v) selacc[v] = 0.f;

        for (int r = 0; r < 5; ++r) {
            v8f c = {};
#pragma unroll
            for (int s = 0; s < 8; ++s) {         // K = 256 in 8 steps of 32
                v16h a;
#pragma unroll
                for (int p = 0; p < 8; ++p) {
                    int kk = (s << 5) + kA(p, g);
                    h2 hv = *(const h2*)&s_h[(i0 + ln) * HP + kk];
                    h2 ar = *(const h2*)&s_ar[r * DIM + kk];
                    h2 av = hv * ar;              // A = h_i * a_r (packed f16)
                    a[2*p] = av.x; a[2*p+1] = av.y;
                }
                c = __builtin_amdgcn_wmma_f32_16x16x32_f16(false, a, false, bf[s],
                                                           (short)0, c, false, false);
            }
#pragma unroll
            for (int v = 0; v < 8; ++v)
                selacc[v] = (adjv[v] == r + 1) ? c[v] : selacc[v];
        }

        // time encoding (only for valid pairs) + leaky relu + mask
#pragma unroll
        for (int v = 0; v < 8; ++v) {
            int ii = i0 + v + (g << 3);
            int r  = adjv[v] - 1;
            float res;
            if (r >= 0) {
                float et = 0.f, av = Ai[v];
                for (int tt = 0; tt < TDIM; ++tt)
                    et += __cosf(av * s_w[tt]) * s_trel[r * TDIM + tt];
                float e = selacc[v] + et;
                res = (e > 0.f) ? e : 0.2f * e;
            } else {
                res = NEGV;
            }
            s_alpha[ii * AP + jj] = res;
        }
    }
    __syncthreads();

    // ---------- Phase 2: row-wise softmax over j ----------
    if (tid < NPAD) {
        float* rowp = s_alpha + tid * AP;
        float mx = -3.4e38f;
        for (int j = 0; j < NPAD; ++j) mx = fmaxf(mx, rowp[j]);
        float sum = 0.f;
        for (int j = 0; j < NPAD; ++j) { float e = __expf(rowp[j] - mx); rowp[j] = e; sum += e; }
        float inv = 1.f / sum;
        for (int j = 0; j < NPAD; ++j) s_ah[tid * AH + j] = (_Float16)(rowp[j] * inv);
    }
    __syncthreads();

    // ---------- Phase 3: out = alpha @ hidden via WMMA ----------
    float* ob = out + (size_t)b * BN * DIM;
    for (int t = wave; t < 64; t += 8) {          // 4 i-tiles x 16 d-tiles
        int i0 = (t >> 4) << 4, d0 = (t & 15) << 4;
        v8f c = {};
#pragma unroll
        for (int s = 0; s < 2; ++s) {             // K = 64 (padded j) in 2 steps
            v16h a, bb;
#pragma unroll
            for (int p = 0; p < 8; ++p) {
                int ka = (s << 5) + kA(p, g);
                int kb = (s << 5) + kB(p, g);
                h2 av = *(const h2*)&s_ah[(i0 + ln) * AH + ka];
                h2 bv = *(const h2*)&s_hT[(d0 + ln) * HTP + kb];  // B[k][n]=h[k][d0+n]
                a[2*p]  = av.x; a[2*p+1]  = av.y;
                bb[2*p] = bv.x; bb[2*p+1] = bv.y;
            }
            c = __builtin_amdgcn_wmma_f32_16x16x32_f16(false, a, false, bb,
                                                       (short)0, c, false, false);
        }
#pragma unroll
        for (int v = 0; v < 8; ++v) {
            int ii = i0 + v + (g << 3);
            if (ii < BN) ob[ii * DIM + d0 + ln] = c[v];
        }
    }
}

extern "C" void kernel_launch(void* const* d_in, const int* in_sizes, int n_in,
                              void* d_out, int out_size, void* d_ws, size_t ws_size,
                              hipStream_t stream) {
    const float* hidden  = (const float*)d_in[0];
    const int*   adjp    = (const int*)d_in[1];
    const float* Aint    = (const float*)d_in[2];
    const float* a_rel   = (const float*)d_in[3];
    const float* t_rel   = (const float*)d_in[4];
    const float* time_w  = (const float*)d_in[5];
    float* out = (float*)d_out;

    int B = in_sizes[0] / (BN * DIM);   // 512

    static_assert(SMEM_BYTES <= 160 * 1024, "LDS budget");
    (void)hipFuncSetAttribute((const void*)tgat_fused_kernel,
                              hipFuncAttributeMaxDynamicSharedMemorySize, SMEM_BYTES);

    tgat_fused_kernel<<<dim3(B), dim3(256), SMEM_BYTES, stream>>>(
        hidden, adjp, Aint, a_rel, t_rel, time_w, out);
}
